// MatrixPairwiseDistances_31181462569006
// MI455X (gfx1250) — compile-verified, matrix operand order checked by reading, not run on hardware
//
#include <hip/hip_runtime.h>

// ---- CDNA5 WMMA / vector types ----
typedef __attribute__((ext_vector_type(16))) __bf16 v16bf;
typedef __attribute__((ext_vector_type(8)))  __bf16 v8bf;
typedef __attribute__((ext_vector_type(4)))  __bf16 v4bf;
typedef __attribute__((ext_vector_type(8)))  float  v8f;
typedef __attribute__((ext_vector_type(4)))  float  v4f;
typedef __attribute__((ext_vector_type(4)))  unsigned int v4u;
typedef __attribute__((ext_vector_type(8)))  int    v8i;
typedef __attribute__((ext_vector_type(4)))  int    v4i;

#define TILE 128
#define DDIM 128
#define LDSB 136   // padded bf16 row stride (272B = 68 dwords: 16B-aligned, 4-dword bank skew)

union Frag { v16bf v; v8bf h[2]; };

// =====================================================================
// Pass 1: f32 -> bf16 hi/lo planes + f32 row norms (one wave per row)
// =====================================================================
__global__ __launch_bounds__(256) void
prep_split_norms(const float* __restrict__ X, const float* __restrict__ Y,
                 __bf16* __restrict__ Xhi, __bf16* __restrict__ Xlo,
                 __bf16* __restrict__ Yhi, __bf16* __restrict__ Ylo,
                 float* __restrict__ xsq, float* __restrict__ ysq,
                 int N, int M)
{
    const int wave = threadIdx.x >> 5, lane = threadIdx.x & 31;
    const int row  = blockIdx.x * 8 + wave;
    if (row >= N + M) return;

    const float* src;
    __bf16 *dhi, *dlo;
    float* nrm;
    if (row < N) { src = X + (size_t)row * DDIM; dhi = Xhi + (size_t)row * DDIM;
                   dlo = Xlo + (size_t)row * DDIM; nrm = xsq + row; }
    else { const int r = row - N;
           src = Y + (size_t)r * DDIM; dhi = Yhi + (size_t)r * DDIM;
           dlo = Ylo + (size_t)r * DDIM; nrm = ysq + r; }

    v4f v = *(const v4f*)(src + lane * 4);
    float s = v.x * v.x + v.y * v.y + v.z * v.z + v.w * v.w;
    v4bf hi, lo;
    #pragma unroll
    for (int j = 0; j < 4; ++j) {
        float f = v[j];
        __bf16 h = (__bf16)f;
        hi[j] = h;
        lo[j] = (__bf16)(f - (float)h);
    }
    *(v4bf*)(dhi + lane * 4) = hi;
    *(v4bf*)(dlo + lane * 4) = lo;
    #pragma unroll
    for (int o = 16; o >= 1; o >>= 1) s += __shfl_xor(s, o, 32);
    if (lane == 0) *nrm = s;
}

// =====================================================================
// TDM helper: DMA one 128x128 bf16 tile global->LDS with HW row padding
// =====================================================================
#if __has_builtin(__builtin_amdgcn_tensor_load_to_lds)
#define HAVE_TDM 1
__device__ __forceinline__ void tdm_load_tile(const __bf16* gsrc, unsigned ldsByteOff)
{
    const unsigned long long ga = (unsigned long long)gsrc;
    v4u g0;
    g0[0] = 1u;                                        // count=1, user mode
    g0[1] = ldsByteOff;                                // lds_addr
    g0[2] = (unsigned)(ga & 0xFFFFFFFFull);            // global_addr[31:0]
    g0[3] = (unsigned)((ga >> 32) & 0x01FFFFFFull)     // global_addr[56:32]
          | (2u << 30);                                // type=2 (image)
    v8i g1;
    g1[0] = (1 << 16)      // data_size = 1 -> 2 bytes
          | (1 << 20)      // pad_enable
          | (5 << 22)      // pad_interval: every 64 dwords (= 256B = one tile row)
          | (3 << 25);     // pad_amount: 4 dwords (16B) -> 272B LDS row stride
    g1[1] = (DDIM & 0xFFFF) << 16;   // tensor_dim0[15:0]   (bits 63:48)
    g1[2] = (TILE & 0xFFFF) << 16;   // tensor_dim1[15:0]   (bits 95:80)
    g1[3] = (TILE & 0xFFFF) << 16;   // tile_dim0           (bits 127:112)
    g1[4] = TILE;                    // tile_dim1           (bits 143:128)
    g1[5] = DDIM;                    // tensor_dim0_stride[31:0] (elements)
    g1[6] = 0;
    g1[7] = 0;
    const v4i z4 = {0, 0, 0, 0};
    const v8i z8 = {0, 0, 0, 0, 0, 0, 0, 0};
    // 6-arg form (clang-23 / therock-10.0 headers): (g0, g1, g2, g3, g4, cpol)
    __builtin_amdgcn_tensor_load_to_lds(g0, g1, z4, z4, z8, 0);
}
#endif

__device__ __forceinline__ void wait_tensorcnt0()
{
#if __has_builtin(__builtin_amdgcn_s_wait_tensorcnt)
    __builtin_amdgcn_s_wait_tensorcnt(0);
#else
    asm volatile("s_wait_tensorcnt 0x0" ::: "memory");
#endif
}

// =====================================================================
// Pass 2: WMMA GEMM over pre-split bf16 planes, TDM-staged LDS tiles
// =====================================================================
__global__ __launch_bounds__(256) void
pairdist_wmma_tdm(const __bf16* __restrict__ Xhi, const __bf16* __restrict__ Xlo,
                  const __bf16* __restrict__ Yhi, const __bf16* __restrict__ Ylo,
                  const float* __restrict__ xsq, const float* __restrict__ ysq,
                  float* __restrict__ Out, int N, int M)
{
    __shared__ __attribute__((aligned(16))) __bf16 sXhi[TILE * LDSB];
    __shared__ __attribute__((aligned(16))) __bf16 sXlo[TILE * LDSB];
    __shared__ __attribute__((aligned(16))) __bf16 sYhi[TILE * LDSB];
    __shared__ __attribute__((aligned(16))) __bf16 sYlo[TILE * LDSB];
    __shared__ float sXsq[TILE];
    __shared__ float sYsq[TILE];

    const int t = threadIdx.x;
    const int wave = t >> 5, lane = t & 31;
    const int xRow0 = blockIdx.y * TILE;   // x rows -> output rows
    const int yRow0 = blockIdx.x * TILE;   // y rows -> output cols

#ifdef HAVE_TDM
    // one DMA per tile, issued by waves 0..3 (EXEC-independent, wave-level op)
    if (wave == 0) tdm_load_tile(Xhi + (size_t)xRow0 * DDIM, (unsigned)(unsigned long long)(void*)sXhi);
    if (wave == 1) tdm_load_tile(Xlo + (size_t)xRow0 * DDIM, (unsigned)(unsigned long long)(void*)sXlo);
    if (wave == 2) tdm_load_tile(Yhi + (size_t)yRow0 * DDIM, (unsigned)(unsigned long long)(void*)sYhi);
    if (wave == 3) tdm_load_tile(Ylo + (size_t)yRow0 * DDIM, (unsigned)(unsigned long long)(void*)sYlo);
#else
    {   // cooperative staging fallback: pure bf16 byte-move into padded LDS
        const __bf16* srcs[4] = { Xhi + (size_t)xRow0 * DDIM, Xlo + (size_t)xRow0 * DDIM,
                                  Yhi + (size_t)yRow0 * DDIM, Ylo + (size_t)yRow0 * DDIM };
        __bf16* dsts[4] = { sXhi, sXlo, sYhi, sYlo };
        const int r = t >> 1, h = t & 1;
        #pragma unroll
        for (int k = 0; k < 4; ++k) {
            const __bf16* s = srcs[k] + (size_t)r * DDIM + h * 64;
            __bf16* d = dsts[k] + r * LDSB + h * 64;
            #pragma unroll
            for (int i = 0; i < 8; ++i) *(v8bf*)(d + 8 * i) = *(const v8bf*)(s + 8 * i);
        }
    }
#endif
    // norms -> LDS (overlaps with TDM DMA)
    if (t < TILE) sXsq[t] = xsq[xRow0 + t];
    else          sYsq[t - TILE] = ysq[yRow0 + (t - TILE)];
    wait_tensorcnt0();
    __syncthreads();

    // ---- WMMA: acc = Xhi*Yhi^T + Xhi*Ylo^T + Xlo*Yhi^T ----
    const int waveM = wave >> 1;          // 0..3 -> 32-row band of x
    const int waveN = wave & 1;           // 0..1 -> 64-col band of y
    const int fr    = lane & 15;
    const int koffL = (lane >> 4) * 8;

    v8f acc[2][4];
    #pragma unroll
    for (int mi = 0; mi < 2; ++mi)
        #pragma unroll
        for (int ni = 0; ni < 4; ++ni)
            acc[mi][ni] = (v8f){0.f, 0.f, 0.f, 0.f, 0.f, 0.f, 0.f, 0.f};

    #pragma unroll
    for (int kb = 0; kb < DDIM; kb += 32) {
        const int ko = kb + koffL;
        Frag ahi[2], alo[2], bhi[4], blo[4];
        #pragma unroll
        for (int mi = 0; mi < 2; ++mi) {
            const int row = (32 * waveM + 16 * mi + fr) * LDSB + ko;
            ahi[mi].h[0] = *(const v8bf*)(sXhi + row);
            ahi[mi].h[1] = *(const v8bf*)(sXhi + row + 16);
            alo[mi].h[0] = *(const v8bf*)(sXlo + row);
            alo[mi].h[1] = *(const v8bf*)(sXlo + row + 16);
        }
        #pragma unroll
        for (int ni = 0; ni < 4; ++ni) {
            const int row = (64 * waveN + 16 * ni + fr) * LDSB + ko;
            bhi[ni].h[0] = *(const v8bf*)(sYhi + row);
            bhi[ni].h[1] = *(const v8bf*)(sYhi + row + 16);
            blo[ni].h[0] = *(const v8bf*)(sYlo + row);
            blo[ni].h[1] = *(const v8bf*)(sYlo + row + 16);
        }
        #pragma unroll
        for (int mi = 0; mi < 2; ++mi)
            #pragma unroll
            for (int ni = 0; ni < 4; ++ni) {
                acc[mi][ni] = __builtin_amdgcn_wmma_f32_16x16x32_bf16(
                    false, ahi[mi].v, false, bhi[ni].v, (short)0, acc[mi][ni], false, false);
                acc[mi][ni] = __builtin_amdgcn_wmma_f32_16x16x32_bf16(
                    false, ahi[mi].v, false, blo[ni].v, (short)0, acc[mi][ni], false, false);
                acc[mi][ni] = __builtin_amdgcn_wmma_f32_16x16x32_bf16(
                    false, alo[mi].v, false, bhi[ni].v, (short)0, acc[mi][ni], false, false);
            }
    }

    // ---- epilogue: d = xsq + ysq - 2*dot, clamped at 0 ----
    const int hiHalf = (lane >> 4) * 8;
    #pragma unroll
    for (int mi = 0; mi < 2; ++mi) {
        #pragma unroll
        for (int ni = 0; ni < 4; ++ni) {
            const int nloc = 64 * waveN + 16 * ni + fr;
            const float yn = sYsq[nloc];
            const size_t gn = (size_t)(yRow0 + nloc);
            #pragma unroll
            for (int r = 0; r < 8; ++r) {
                const int mloc = 32 * waveM + 16 * mi + r + hiHalf;
                const float d = fmaf(-2.0f, acc[mi][ni][r], sXsq[mloc] + yn);
                Out[(size_t)(xRow0 + mloc) * (size_t)M + gn] = fmaxf(d, 0.0f);
            }
        }
    }
}

// =====================================================================
// Fallback (ws too small): self-contained round-1 kernel
// =====================================================================
__global__ __launch_bounds__(256) void
pairdist_wmma_selfcontained(const float* __restrict__ X, const float* __restrict__ Y,
                            float* __restrict__ Out, int N, int M)
{
    __shared__ __attribute__((aligned(16))) __bf16 sXhi[TILE * LDSB];
    __shared__ __attribute__((aligned(16))) __bf16 sXlo[TILE * LDSB];
    __shared__ __attribute__((aligned(16))) __bf16 sYhi[TILE * LDSB];
    __shared__ __attribute__((aligned(16))) __bf16 sYlo[TILE * LDSB];
    __shared__ float sXsqp[2 * TILE];
    __shared__ float sYsqp[2 * TILE];

    const int t = threadIdx.x;
    const int xRow0 = blockIdx.y * TILE;
    const int yRow0 = blockIdx.x * TILE;
    {
        const int r = t >> 1, h = t & 1;
        const float* srcX = X + (size_t)(xRow0 + r) * DDIM + h * 64;
        const float* srcY = Y + (size_t)(yRow0 + r) * DDIM + h * 64;
        __bf16* dXh = sXhi + r * LDSB + h * 64;  __bf16* dXl = sXlo + r * LDSB + h * 64;
        __bf16* dYh = sYhi + r * LDSB + h * 64;  __bf16* dYl = sYlo + r * LDSB + h * 64;
        float ssqx = 0.f, ssqy = 0.f;
        #pragma unroll
        for (int i = 0; i < 16; ++i) {
            v4f vx = *(const v4f*)(srcX + 4 * i);
            v4f vy = *(const v4f*)(srcY + 4 * i);
            ssqx += vx.x * vx.x + vx.y * vx.y + vx.z * vx.z + vx.w * vx.w;
            ssqy += vy.x * vy.x + vy.y * vy.y + vy.z * vy.z + vy.w * vy.w;
            v4bf xh, xl, yh, yl;
            #pragma unroll
            for (int j = 0; j < 4; ++j) {
                float fx = vx[j], fy = vy[j];
                __bf16 hx = (__bf16)fx;  xh[j] = hx;  xl[j] = (__bf16)(fx - (float)hx);
                __bf16 hy = (__bf16)fy;  yh[j] = hy;  yl[j] = (__bf16)(fy - (float)hy);
            }
            *(v4bf*)(dXh + 4 * i) = xh;  *(v4bf*)(dXl + 4 * i) = xl;
            *(v4bf*)(dYh + 4 * i) = yh;  *(v4bf*)(dYl + 4 * i) = yl;
        }
        sXsqp[t] = ssqx;  sYsqp[t] = ssqy;
    }
    __syncthreads();

    const int wave = t >> 5, lane = t & 31;
    const int waveM = wave >> 1, waveN = wave & 1;
    const int fr = lane & 15, koffL = (lane >> 4) * 8;
    v8f acc[2][4];
    #pragma unroll
    for (int mi = 0; mi < 2; ++mi)
        #pragma unroll
        for (int ni = 0; ni < 4; ++ni)
            acc[mi][ni] = (v8f){0.f, 0.f, 0.f, 0.f, 0.f, 0.f, 0.f, 0.f};
    #pragma unroll
    for (int kb = 0; kb < DDIM; kb += 32) {
        const int ko = kb + koffL;
        Frag ahi[2], alo[2], bhi[4], blo[4];
        #pragma unroll
        for (int mi = 0; mi < 2; ++mi) {
            const int row = (32 * waveM + 16 * mi + fr) * LDSB + ko;
            ahi[mi].h[0] = *(const v8bf*)(sXhi + row);  ahi[mi].h[1] = *(const v8bf*)(sXhi + row + 16);
            alo[mi].h[0] = *(const v8bf*)(sXlo + row);  alo[mi].h[1] = *(const v8bf*)(sXlo + row + 16);
        }
        #pragma unroll
        for (int ni = 0; ni < 4; ++ni) {
            const int row = (64 * waveN + 16 * ni + fr) * LDSB + ko;
            bhi[ni].h[0] = *(const v8bf*)(sYhi + row);  bhi[ni].h[1] = *(const v8bf*)(sYhi + row + 16);
            blo[ni].h[0] = *(const v8bf*)(sYlo + row);  blo[ni].h[1] = *(const v8bf*)(sYlo + row + 16);
        }
        #pragma unroll
        for (int mi = 0; mi < 2; ++mi)
            #pragma unroll
            for (int ni = 0; ni < 4; ++ni) {
                acc[mi][ni] = __builtin_amdgcn_wmma_f32_16x16x32_bf16(
                    false, ahi[mi].v, false, bhi[ni].v, (short)0, acc[mi][ni], false, false);
                acc[mi][ni] = __builtin_amdgcn_wmma_f32_16x16x32_bf16(
                    false, ahi[mi].v, false, blo[ni].v, (short)0, acc[mi][ni], false, false);
                acc[mi][ni] = __builtin_amdgcn_wmma_f32_16x16x32_bf16(
                    false, alo[mi].v, false, bhi[ni].v, (short)0, acc[mi][ni], false, false);
            }
    }
    const int hiHalf = (lane >> 4) * 8;
    #pragma unroll
    for (int mi = 0; mi < 2; ++mi)
        #pragma unroll
        for (int ni = 0; ni < 4; ++ni) {
            const int nloc = 64 * waveN + 16 * ni + fr;
            const float yn = sYsqp[2 * nloc] + sYsqp[2 * nloc + 1];
            const size_t gn = (size_t)(yRow0 + nloc);
            #pragma unroll
            for (int r = 0; r < 8; ++r) {
                const int mloc = 32 * waveM + 16 * mi + r + hiHalf;
                const float xs = sXsqp[2 * mloc] + sXsqp[2 * mloc + 1];
                const float d = fmaf(-2.0f, acc[mi][ni][r], xs + yn);
                Out[(size_t)(xRow0 + mloc) * (size_t)M + gn] = fmaxf(d, 0.0f);
            }
        }
}

extern "C" void kernel_launch(void* const* d_in, const int* in_sizes, int n_in,
                              void* d_out, int out_size, void* d_ws, size_t ws_size,
                              hipStream_t stream) {
    const float* x = (const float*)d_in[0];
    const float* y = (const float*)d_in[1];
    float* out = (float*)d_out;
    const int D = 128;
    const int N = in_sizes[0] / D;   // 8192
    const int M = in_sizes[1] / D;   // 8192
    dim3 grid(M / TILE, N / TILE);

    const size_t need = (size_t)(N + M) * D * 2 /*hi+lo*/ * sizeof(__bf16)
                      + (size_t)(N + M) * sizeof(float);
    if (ws_size >= need) {
        __bf16* Xhi = (__bf16*)d_ws;
        __bf16* Xlo = Xhi + (size_t)N * D;
        __bf16* Yhi = Xlo + (size_t)N * D;
        __bf16* Ylo = Yhi + (size_t)M * D;
        float*  xsq = (float*)(Ylo + (size_t)M * D);
        float*  ysq = xsq + N;
        const int rows = N + M;
        prep_split_norms<<<(rows + 7) / 8, 256, 0, stream>>>(x, y, Xhi, Xlo, Yhi, Ylo,
                                                             xsq, ysq, N, M);
        pairdist_wmma_tdm<<<grid, 256, 0, stream>>>(Xhi, Xlo, Yhi, Ylo, xsq, ysq, out, N, M);
    } else {
        pairdist_wmma_selfcontained<<<grid, 256, 0, stream>>>(x, y, out, N, M);
    }
}
